// StageNet_16913581212072
// MI455X (gfx1250) — compile-verified, matrix operand order checked by reading, not run on hardware
//
#include <hip/hip_runtime.h>
#include <hip/hip_bf16.h>

// ---------------------------------------------------------------- constants
constexpr int BB   = 1024;            // batch
constexpr int TS   = 256;             // timesteps
constexpr int DIN  = 256;             // input dim
constexpr int HID  = 768;             // hidden dim
constexpr int LVL  = 6;               // levels
constexpr int KW   = 10;              // conv window
constexpr int GD   = 4 * HID + 2 * LVL;   // 3084 gate width
constexpr int GDP  = 3104;            // padded to multiple of 32 (wave N-tile)
constexpr int KA   = DIN + HID;       // 1024 total K per step

typedef __attribute__((ext_vector_type(16))) __bf16 v16bf;
typedef __attribute__((ext_vector_type(8)))  float  v8f;

// ---------------------------------------------------------------- helpers
__device__ __forceinline__ float sigf(float v) { return 1.f / (1.f + __expf(-v)); }

__device__ __forceinline__ v8f wmma_bf16(v16bf a, v16bf b, v8f c) {
    return __builtin_amdgcn_wmma_f32_16x16x32_bf16(
        /*neg_a=*/false, a, /*neg_b=*/false, b,
        /*c_mod=*/(short)0, c, /*reuse_a=*/false, /*reuse_b=*/false);
}

// Swizzled LDS addressing: rows are 1024 bf16 (2048 B); we XOR the 16-byte
// chunk index with (row & 15) inside each 256 B group, so 16 lanes reading 16
// different rows at the same K hit 16 distinct chunks = all 64 banks.
__device__ __forceinline__ int lds_idx(int r, int k) {
    const int c  = k >> 3;                       // 16-byte chunk index
    const int cs = (c & ~15) | ((c ^ r) & 15);   // swizzle within 256B group
    return r * KA + (cs << 3) + (k & 7);
}

// A fragment (16x32, 16-bit) from the swizzled LDS tile, rows rbase..rbase+15:
// lanes 0-15 -> row M=lane, K in {0..7,16..23}; lanes 16-31 -> K in {8..15,24..31}.
__device__ __forceinline__ v16bf load_a_lds(const __bf16* As, int rbase, int kk, int lane) {
    const int r  = rbase + (lane & 15);
    const int k0 = kk + ((lane >> 4) << 3);      // multiple of 8
    const __bf16* p0 = As + lds_idx(r, k0);
    const __bf16* p1 = As + lds_idx(r, k0 + 16);
    v16bf a;
#pragma unroll
    for (int i = 0; i < 8; ++i) a[i] = p0[i];
#pragma unroll
    for (int i = 0; i < 8; ++i) a[8 + i] = p1[i];
    return a;
}

// A fragment straight from fp32 global (used by tail GEMMs).
__device__ __forceinline__ v16bf load_a_f32(const float* __restrict__ base, long ld,
                                            int row0, int kk, int lane) {
    const int m = row0 + (lane & 15);
    const float* p = base + (long)m * ld + kk + ((lane >> 4) << 3);
    v16bf a;
#pragma unroll
    for (int i = 0; i < 8; ++i) a[i] = (__bf16)p[i];
#pragma unroll
    for (int i = 0; i < 8; ++i) a[8 + i] = (__bf16)p[16 + i];
    return a;
}

// B fragment (32x16, 16-bit) for C = A @ W^T: W row-major [N, K] in bf16.
__device__ __forceinline__ v16bf load_b_bf16(const __bf16* __restrict__ W, long ld,
                                             int col0, int kk, int lane) {
    const int n = col0 + (lane & 15);
    const __bf16* p = W + (long)n * ld + kk + ((lane >> 4) << 3);
    v16bf b;
#pragma unroll
    for (int i = 0; i < 8; ++i) b[i] = p[i];
#pragma unroll
    for (int i = 0; i < 8; ++i) b[8 + i] = p[16 + i];
    return b;
}

// ------------------------------------------------------- per-step gate GEMM
// g[B, GD] = x_t[B, DIN] @ Wx^T + h[B, HID] @ Wh^T   (bias / time handled later)
// block = 8 waves; 32 A rows staged once in LDS (bf16, swizzled, exactly 64 KB).
// Each wave owns a 32x32 output tile: 2 A frags x 2 B frags -> 4 wmma per K-chunk.
__global__ __launch_bounds__(256) void step_gemm_kernel(
    const float*  __restrict__ x_t, long ldx,     // fp32 x slice, row stride TS*DIN
    const __bf16* __restrict__ hbf,               // [BB, HID] bf16 hidden state
    const __bf16* __restrict__ Wx,                // [GDP, DIN] bf16 (pad rows zeroed)
    const __bf16* __restrict__ Wh,                // [GDP, HID] bf16
    float* __restrict__ g)
{
    __shared__ __bf16 As[32 * KA];                // 65,536 B (of 320 KB/WGP)
    const int tid  = threadIdx.x;
    const int row0 = blockIdx.y * 32;

    // stage x rows (fp32 -> bf16, converted ONCE per block)
    for (int i = tid; i < 32 * DIN; i += 256) {
        const int r = i >> 8, k = i & (DIN - 1);
        As[lds_idx(r, k)] = (__bf16)x_t[(long)(row0 + r) * ldx + k];
    }
    // stage h rows (already bf16 — plain copy)
    for (int i = tid; i < 32 * HID; i += 256) {
        const int r = i / HID, k = i - r * HID;
        As[lds_idx(r, DIN + k)] = hbf[(long)(row0 + r) * HID + k];
    }
    __syncthreads();

    const int lane = tid & 31;
    const int wave = tid >> 5;
    const int col0 = (blockIdx.x * 8 + wave) * 32;
    if (col0 >= GDP) return;                      // uniform per wave, after barrier

    v8f acc[2][2];
#pragma unroll
    for (int mi = 0; mi < 2; ++mi)
#pragma unroll
        for (int ni = 0; ni < 2; ++ni)
            acc[mi][ni] = (v8f){0.f, 0.f, 0.f, 0.f, 0.f, 0.f, 0.f, 0.f};

#pragma unroll 2
    for (int kk = 0; kk < DIN; kk += 32) {
        const v16bf a0 = load_a_lds(As, 0,  kk, lane);
        const v16bf a1 = load_a_lds(As, 16, kk, lane);
        const v16bf b0 = load_b_bf16(Wx, DIN, col0,      kk, lane);
        const v16bf b1 = load_b_bf16(Wx, DIN, col0 + 16, kk, lane);
        acc[0][0] = wmma_bf16(a0, b0, acc[0][0]);
        acc[0][1] = wmma_bf16(a0, b1, acc[0][1]);
        acc[1][0] = wmma_bf16(a1, b0, acc[1][0]);
        acc[1][1] = wmma_bf16(a1, b1, acc[1][1]);
    }
#pragma unroll 2
    for (int kk = 0; kk < HID; kk += 32) {
        if (kk + 32 < HID)   // keep the L2-resident weight stream ahead
            __builtin_prefetch(Wh + (long)(col0 + (lane & 15)) * HID + kk + 32, 0, 1);
        const v16bf a0 = load_a_lds(As, 0,  DIN + kk, lane);
        const v16bf a1 = load_a_lds(As, 16, DIN + kk, lane);
        const v16bf b0 = load_b_bf16(Wh, HID, col0,      kk, lane);
        const v16bf b1 = load_b_bf16(Wh, HID, col0 + 16, kk, lane);
        acc[0][0] = wmma_bf16(a0, b0, acc[0][0]);
        acc[0][1] = wmma_bf16(a0, b1, acc[0][1]);
        acc[1][0] = wmma_bf16(a1, b0, acc[1][0]);
        acc[1][1] = wmma_bf16(a1, b1, acc[1][1]);
    }

    const int rbase = (lane >> 4) << 3;
#pragma unroll
    for (int mi = 0; mi < 2; ++mi)
#pragma unroll
        for (int ni = 0; ni < 2; ++ni) {
            const int cc = col0 + ni * 16 + (lane & 15);
            if (cc < GD) {
#pragma unroll
                for (int r = 0; r < 8; ++r) {
                    const int rr = row0 + mi * 16 + r + rbase;
                    g[(long)rr * GD + cc] = acc[mi][ni][r];
                }
            }
        }
}

// --------------------------------------------- generic tail GEMM + epilogue
// C[M,N] = epi(A[M,K] @ W[N,K]^T + bias)   epi: 1 = bias, 2 = bias+relu, 3 = bias+sigmoid
__global__ __launch_bounds__(256) void gemm_epi_kernel(
    const float* __restrict__ A, long lda,
    const __bf16* __restrict__ W, long ldw,
    const float* __restrict__ bias,
    float* __restrict__ C, long ldc,
    int N, int Ktot, int epi)
{
    const int lane = threadIdx.x & 31;
    const int wave = threadIdx.x >> 5;
    const int col0 = (blockIdx.x * 8 + wave) * 16;
    const int row0 = blockIdx.y * 16;
    if (col0 >= N) return;

    v8f acc = {0.f, 0.f, 0.f, 0.f, 0.f, 0.f, 0.f, 0.f};
    for (int kk = 0; kk < Ktot; kk += 32) {
        v16bf a = load_a_f32(A, lda, row0, kk, lane);
        v16bf b = load_b_bf16(W, ldw, col0, kk, lane);
        acc = wmma_bf16(a, b, acc);
    }

    const int cc = col0 + (lane & 15);
    const float bv = bias[cc];
#pragma unroll
    for (int r = 0; r < 8; ++r) {
        const int rr = row0 + r + ((lane >> 4) << 3);
        float v = acc[r] + bv;
        if (epi == 2) v = v > 0.f ? v : 0.f;
        else if (epi == 3) v = sigf(v);
        C[(long)rr * ldc + cc] = v;
    }
}

// ------------------------------------------------- per-step gate/cell update
__global__ __launch_bounds__(256) void gate_cell_kernel(
    const float* __restrict__ g, const float* __restrict__ time,
    const float* __restrict__ biasE, const float* __restrict__ tcol,
    float* __restrict__ h, __bf16* __restrict__ hbf, float* __restrict__ c,
    float* __restrict__ hwin, float* __restrict__ dwin,
    float* __restrict__ dist_out, int t)
{
    const int b = blockIdx.x;
    const int tid = threadIdx.x;
    __shared__ float fm[LVL], im[LVL];
    const float tt = time[(long)b * TS + t];

    if (tid == 0) {
        float mg[2 * LVL];
#pragma unroll
        for (int j = 0; j < 2 * LVL; ++j)
            mg[j] = g[(long)b * GD + j] + biasE[j] + tt * tcol[j];
        // f_m = cumsum(softmax(mg[0:6]))  (left-to-right)
        float mx = mg[0];
        for (int l = 1; l < LVL; ++l) mx = fmaxf(mx, mg[l]);
        float e[LVL], s = 0.f;
        for (int l = 0; l < LVL; ++l) { e[l] = __expf(mg[l] - mx); s += e[l]; }
        float cum = 0.f, mean = 0.f;
        for (int l = 0; l < LVL; ++l) { cum += e[l] / s; fm[l] = cum; mean += cum; }
        // i_m: flip/softmax/cumsum/flip == suffix-sum of softmax(mg[6:12])
        float mx2 = mg[LVL];
        for (int l = 1; l < LVL; ++l) mx2 = fmaxf(mx2, mg[LVL + l]);
        float e2[LVL], s2 = 0.f;
        for (int l = 0; l < LVL; ++l) { e2[l] = __expf(mg[LVL + l] - mx2); s2 += e2[l]; }
        float cr = 0.f;
        for (int l = LVL - 1; l >= 0; --l) { cr += e2[l] / s2; im[l] = cr; }

        const float dist = 1.f - mean / (float)LVL;
        dwin[(long)(t % KW) * BB + b] = dist;
        dist_out[(long)t * BB + b]   = dist;
    }
    __syncthreads();

    const long base = (long)b * GD + 2 * LVL;
    for (int ch = tid; ch < HID; ch += 256) {
        const int l = ch >> 7;                      // CH = 128
        const float fg = sigf (g[base + ch]         + biasE[2*LVL + ch]         + tt * tcol[2*LVL + ch]);
        const float ig = sigf (g[base + HID + ch]   + biasE[2*LVL + HID + ch]   + tt * tcol[2*LVL + HID + ch]);
        const float og = sigf (g[base + 2*HID + ch] + biasE[2*LVL + 2*HID + ch] + tt * tcol[2*LVL + 2*HID + ch]);
        const float ci = tanhf(g[base + 3*HID + ch] + biasE[2*LVL + 3*HID + ch] + tt * tcol[2*LVL + 3*HID + ch]);
        const float cl = c[(long)b * HID + ch];
        const float f = fm[l], i = im[l], ov = f * i;
        const float cn = ov * (fg * cl + ig * ci) + (f - ov) * cl + (i - ov) * ci;
        const float hn = og * tanhf(cn);
        c[(long)b * HID + ch]   = cn;
        h[(long)b * HID + ch]   = hn;          // fp32 copy for the final residual
        hbf[(long)b * HID + ch] = (__bf16)hn;  // bf16 copy feeding next step's WMMA
        hwin[((long)(t % KW) * BB + b) * HID + ch] = hn;
    }
}

// -------------------------------------------------------- window / pooling
__global__ __launch_bounds__(256) void window_kernel(
    const float* __restrict__ hwin, const float* __restrict__ dwin,
    float* __restrict__ local_h, float* __restrict__ pooled)
{
    const int b = blockIdx.x;
    const int tid = threadIdx.x;
    __shared__ float ld[KW];
    if (tid == 0) {
        float cs = 0.f, v[KW], mx = -1e30f;
        for (int w = 0; w < KW; ++w) {
            const int slot = (TS - KW + w) % KW;      // oldest -> newest
            cs += dwin[(long)slot * BB + b];
            v[w] = cs;
            mx = fmaxf(mx, cs);
        }
        float s = 0.f;
        for (int w = 0; w < KW; ++w) { v[w] = __expf(v[w] - mx); s += v[w]; }
        for (int w = 0; w < KW; ++w) ld[w] = v[w] / s;
    }
    __syncthreads();
    for (int ch = tid; ch < HID; ch += 256) {
        float acc = 0.f;
#pragma unroll
        for (int w = 0; w < KW; ++w) {
            const int slot = (TS - KW + w) % KW;
            const float val = hwin[((long)slot * BB + b) * HID + ch] * ld[w];
            local_h[((long)b * HID + ch) * KW + w] = val;
            acc += val;
        }
        pooled[(long)b * HID + ch] = acc / (float)KW;
    }
}

// ----------------------------------------------------------------- finalize
__global__ __launch_bounds__(256) void final_kernel(
    const float* __restrict__ theme, const float* __restrict__ convO,
    const float* __restrict__ h, const float* __restrict__ out_w,
    const float* __restrict__ out_b, float* __restrict__ out)
{
    const int b = blockIdx.x;
    const int tid = threadIdx.x;
    __shared__ float red[256];
    float s = 0.f;
    for (int ch = tid; ch < HID; ch += 256) {
        const long i = (long)b * HID + ch;
        s += (theme[i] * convO[i] + h[i]) * out_w[ch];
    }
    red[tid] = s;
    __syncthreads();
    for (int off = 128; off > 0; off >>= 1) {
        if (tid < off) red[tid] += red[tid + off];
        __syncthreads();
    }
    if (tid == 0) out[b] = sigf(red[0] + out_b[0]);
}

// ------------------------------------------------------------- prep kernels
__global__ void cvt_f32_bf16_kernel(const float* __restrict__ src, long src_ld, int cols,
                                    __bf16* __restrict__ dst, int rows, long n) {
    const long i = (long)blockIdx.x * 256 + threadIdx.x;
    if (i >= n) return;
    const long r = i / cols, cc = i % cols;
    dst[i] = (r < rows) ? (__bf16)src[r * src_ld + cc] : (__bf16)0.f;
}

__global__ void prep_cols_kernel(const float* __restrict__ kw, const float* __restrict__ kb,
                                 const float* __restrict__ rw, const float* __restrict__ rb,
                                 float* __restrict__ biasE, float* __restrict__ tcol) {
    const int j = blockIdx.x * 256 + threadIdx.x;
    if (j < GD) {
        biasE[j] = kb[j] + rb[j];
        tcol[j]  = kw[(long)j * (DIN + 1) + DIN] + rw[(long)j * (HID + 1) + HID];
    }
}

__global__ void zero_kernel(float* __restrict__ p, long n) {
    const long i = (long)blockIdx.x * 256 + threadIdx.x;
    if (i < n) p[i] = 0.f;
}

__global__ void zero_bf16_kernel(__bf16* __restrict__ p, long n) {
    const long i = (long)blockIdx.x * 256 + threadIdx.x;
    if (i < n) p[i] = (__bf16)0.f;
}

// ----------------------------------------------------------------- launcher
extern "C" void kernel_launch(void* const* d_in, const int* in_sizes, int n_in,
                              void* d_out, int out_size, void* d_ws, size_t ws_size,
                              hipStream_t stream) {
    const float* x         = (const float*)d_in[0];
    const float* timep     = (const float*)d_in[1];
    const float* kernel_w  = (const float*)d_in[2];
    const float* kernel_b  = (const float*)d_in[3];
    const float* rec_w     = (const float*)d_in[4];
    const float* rec_b     = (const float*)d_in[5];
    const float* scale_w   = (const float*)d_in[6];
    const float* scale_b   = (const float*)d_in[7];
    const float* rescale_w = (const float*)d_in[8];
    const float* rescale_b = (const float*)d_in[9];
    const float* conv_w    = (const float*)d_in[10];
    const float* conv_b    = (const float*)d_in[11];
    const float* out_w     = (const float*)d_in[12];
    const float* out_b     = (const float*)d_in[13];

    float* out      = (float*)d_out;        // [B]
    float* dist_out = out + BB;             // [T, B]

    // ---- workspace carve-out (aligned to 256B)
    char* ws = (char*)d_ws;
    auto carve = [&](size_t bytes) {
        void* p = (void*)ws;
        ws += (bytes + 255) & ~(size_t)255;
        return p;
    };
    __bf16* WxB   = (__bf16*)carve((size_t)GDP * DIN * 2);
    __bf16* WhB   = (__bf16*)carve((size_t)GDP * HID * 2);
    __bf16* SwB   = (__bf16*)carve((size_t)(HID / 6) * HID * 2);
    __bf16* RwB   = (__bf16*)carve((size_t)HID * (HID / 6) * 2);
    __bf16* CwB   = (__bf16*)carve((size_t)HID * HID * KW * 2);
    __bf16* hbfB  = (__bf16*)carve((size_t)BB * HID * 2);
    float*  biasE = (float*)carve((size_t)GD * 4);
    float*  tcol  = (float*)carve((size_t)GD * 4);
    float*  gbuf  = (float*)carve((size_t)BB * GD * 4);
    float*  hbuf  = (float*)carve((size_t)BB * HID * 4);
    float*  cbuf  = (float*)carve((size_t)BB * HID * 4);
    float*  hwin  = (float*)carve((size_t)KW * BB * HID * 4);
    float*  dwin  = (float*)carve((size_t)KW * BB * 4);
    float*  lochB = (float*)carve((size_t)BB * HID * KW * 4);
    float*  pooled= (float*)carve((size_t)BB * HID * 4);
    float*  s1    = (float*)carve((size_t)BB * (HID / 6) * 4);
    float*  theme = (float*)carve((size_t)BB * HID * 4);
    float*  convO = (float*)carve((size_t)BB * HID * 4);

    // ---- one-time (per call) conversions & prep
    {
        long n;
        n = (long)GDP * DIN;
        cvt_f32_bf16_kernel<<<(n + 255) / 256, 256, 0, stream>>>(kernel_w, DIN + 1, DIN, WxB, GD, n);
        n = (long)GDP * HID;
        cvt_f32_bf16_kernel<<<(n + 255) / 256, 256, 0, stream>>>(rec_w, HID + 1, HID, WhB, GD, n);
        n = (long)(HID / 6) * HID;
        cvt_f32_bf16_kernel<<<(n + 255) / 256, 256, 0, stream>>>(scale_w, HID, HID, SwB, HID / 6, n);
        n = (long)HID * (HID / 6);
        cvt_f32_bf16_kernel<<<(n + 255) / 256, 256, 0, stream>>>(rescale_w, HID / 6, HID / 6, RwB, HID, n);
        n = (long)HID * HID * KW;
        cvt_f32_bf16_kernel<<<(n + 255) / 256, 256, 0, stream>>>(conv_w, HID * KW, HID * KW, CwB, HID, n);
        prep_cols_kernel<<<(GD + 255) / 256, 256, 0, stream>>>(kernel_w, kernel_b, rec_w, rec_b, biasE, tcol);
        n = (long)BB * HID;
        zero_kernel<<<(n + 255) / 256, 256, 0, stream>>>(hbuf, n);
        zero_kernel<<<(n + 255) / 256, 256, 0, stream>>>(cbuf, n);
        zero_bf16_kernel<<<(n + 255) / 256, 256, 0, stream>>>(hbfB, n);
    }

    // ---- recurrent scan: 256 sequential steps (WMMA GEMM + gate/cell)
    // 8 waves/block, each wave computes a 32x32 tile -> block covers 256 cols.
    const dim3 gemm_grid((GDP / 32 + 7) / 8, BB / 32);   // 13 x 32
    for (int t = 0; t < TS; ++t) {
        step_gemm_kernel<<<gemm_grid, 256, 0, stream>>>(
            x + (long)t * DIN, (long)TS * DIN, hbfB, WxB, WhB, gbuf);
        gate_cell_kernel<<<BB, 256, 0, stream>>>(
            gbuf, timep, biasE, tcol, hbuf, hbfB, cbuf, hwin, dwin, dist_out, t);
    }

    // ---- tail
    window_kernel<<<BB, 256, 0, stream>>>(hwin, dwin, lochB, pooled);
    // theme1: relu(pooled @ scale_w^T + scale_b)   [1024,768]x[128,768]^T
    gemm_epi_kernel<<<dim3(1, BB / 16), 256, 0, stream>>>(
        pooled, HID, SwB, HID, scale_b, s1, HID / 6, HID / 6, HID, 2);
    // theme2: sigmoid(s1 @ rescale_w^T + rescale_b) [1024,128]x[768,128]^T
    gemm_epi_kernel<<<dim3(6, BB / 16), 256, 0, stream>>>(
        s1, HID / 6, RwB, HID / 6, rescale_b, theme, HID, HID, HID / 6, 3);
    // conv: local_h @ conv_w^T + conv_b            [1024,7680]x[768,7680]^T
    gemm_epi_kernel<<<dim3(6, BB / 16), 256, 0, stream>>>(
        lochB, (long)HID * KW, CwB, (long)HID * KW, conv_b, convO, HID, HID, HID * KW, 1);
    final_kernel<<<BB, 256, 0, stream>>>(theme, convO, hbuf, out_w, out_b, out);
}
